// RVIN_83373905150378
// MI455X (gfx1250) — matrix operand order repaired and checked by reading, not compiled, not run on hardware
//
#include <hip/hip_runtime.h>

// ---------------------------------------------------------------------------
// MI455X (gfx1250, wave32) implementation of a VIN + LSTM head.
// The h and VI q convs run as implicit GEMM on v_wmma_f32_16x16x32_f16 (f16 in, f32 acc).
// K-slot permutation: within each 32-wide K-step, lane-half selects the input
// channel and element index (<9) selects the 3x3 tap, so B-fragment builds are
// 9 constant-offset LDS loads from one per-lane base pointer.
// h chunk is stored channel-last so stores are b128 and the h->r reduction
// reads 16 channels per tap as one 32B vector load.
// ---------------------------------------------------------------------------

typedef __attribute__((ext_vector_type(16))) _Float16 v16h;
typedef __attribute__((ext_vector_type(8)))  _Float16 v8h;
typedef __attribute__((ext_vector_type(8)))  float    v8f;

__device__ __forceinline__ v8f wmma_f16(v16h a, v16h b, v8f c) {
    return __builtin_amdgcn_wmma_f32_16x16x32_f16(
        false, a, false, b, (short)0, c, false, false);
}

// A-matrix (16x32 f16) K slot for vector element e on lane `ln` (this code's contiguous order, 16*half + e)
__device__ __forceinline__ int a_kidx(int ln, int e) {
    int half = ln >> 4;
    return half * 16 + e;
}

__device__ __forceinline__ float sigf(float v) { return 1.0f / (1.0f + __expf(-v)); }

// ---------------------------------------------------------------------------
// Kernel 1: per-image flat index of first x[:,0] > 0 (jnp.argmax on bool)
// ---------------------------------------------------------------------------
__global__ void vin_pos_kernel(const float* __restrict__ x, int* __restrict__ p_out) {
    int b = blockIdx.x;
    const float* xc = x + (size_t)b * 4 * 4096;
    __shared__ int smin;
    if (threadIdx.x == 0) smin = 4096;
    __syncthreads();
    int best = 4096;
    for (int i = threadIdx.x; i < 4096; i += blockDim.x)
        if (xc[i] > 0.0f && i < best) best = i;
    atomicMin(&smin, best);
    __syncthreads();
    if (threadIdx.x == 0) p_out[b] = (smin == 4096) ? 0 : smin;
}

// ---------------------------------------------------------------------------
// Kernel 2: fused h = conv(x,Wh)+bh ; r = conv(h,Wr)+br
// grid = (4 row-bands, 64 images), 256 threads (8 waves).
// ---------------------------------------------------------------------------
__global__ void vin_hr_kernel(const float* __restrict__ x,
                              const float* __restrict__ Wh,
                              const float* __restrict__ bh,
                              const float* __restrict__ Wr,
                              const float* __restrict__ br,
                              float* __restrict__ r_out) {
    const int band = blockIdx.x;
    const int b    = blockIdx.y;
    const int y0   = band * 16;
    const int tid  = threadIdx.x;
    const int ln   = tid & 31;
    const int wv   = __builtin_amdgcn_readfirstlane(tid >> 5);  // scalar wave id
    const int half = ln >> 4;

    __shared__ _Float16 xs[4][20][72];                   // x band, rows y0-2..y0+17, cols -1..64
    __shared__ __align__(32) _Float16 wA[10][2][32][16]; // Wh A fragments
    __shared__ __align__(32) _Float16 hs[18][68][16];    // h chunk, channel-last
    __shared__ float    wrsT[9][160];                    // Wr transposed [tap][channel]
    __shared__ float    bhs[160];

    for (int idx = tid; idx < 4 * 20 * 72; idx += 256) {
        int col = idx % 72, row = (idx / 72) % 20, c = idx / (72 * 20);
        int gy = y0 - 2 + row, gx = col - 1;
        float v = 0.0f;
        if (gy >= 0 && gy < 64 && gx >= 0 && gx < 64)
            v = x[(((size_t)b * 4 + c) * 64 + gy) * 64 + gx];
        xs[c][row][col] = (_Float16)v;
    }
    // A fragments under the permuted K mapping
    for (int idx = tid; idx < 10240; idx += 256) {
        int e = idx & 15, lane = (idx >> 4) & 31, kstep = (idx >> 9) & 1, chunk = idx >> 10;
        int o   = chunk * 16 + (lane & 15);
        int khw = a_kidx(lane, e);            // K slot 0..31 in this K-step
        int c   = kstep * 2 + (khw >> 4);     // logical channel
        int s   = khw & 15;                   // logical 3x3 tap
        float w = 0.0f;
        if (o < 150 && s < 9)
            w = Wh[((o * 4 + c) * 3 + s / 3) * 3 + s % 3];
        wA[chunk][kstep][lane][e] = (_Float16)w;
    }
    for (int idx = tid; idx < 160 * 9; idx += 256) {
        int o = idx / 9, t = idx % 9;
        wrsT[t][o] = (o < 150) ? Wr[o * 9 + t] : 0.0f;
    }
    for (int o = tid; o < 160; o += 256) bhs[o] = (o < 150) ? bh[o] : 0.0f;
    for (int idx = tid; idx < 18 * 16; idx += 256) {     // column halo of h
        int row = idx / 16, m = idx % 16;
        hs[row][0][m]  = (_Float16)0.0f;
        hs[row][65][m] = (_Float16)0.0f;
    }

    float racc[4] = {0.f, 0.f, 0.f, 0.f};
    __syncthreads();

    const int nlo = ln & 15;
    for (int chunk = 0; chunk < 10; ++chunk) {
        v16h a0 = *(const v16h*)&wA[chunk][0][ln][0];
        v16h a1 = *(const v16h*)&wA[chunk][1][ln][0];
        v8f cinit;
        #pragma unroll
        for (int v = 0; v < 8; ++v) cinit[v] = bhs[chunk * 16 + v + half * 8];

        #pragma unroll 3
        for (int t0 = 0; t0 < 9; ++t0) {                 // 72 tiles, 9 per wave
            int tl  = wv + t0 * 8;
            int pos = tl * 16 + nlo;
            int hr  = pos >> 6, hc = pos & 63;
            v8f c8 = cinit;
            #pragma unroll
            for (int kstep = 0; kstep < 2; ++kstep) {
                int c = kstep * 2 + half;                // channel from lane-half
                const _Float16* base = &xs[c][hr + 1][hc + 1];
                v16h bf;
                #pragma unroll
                for (int s = 0; s < 9; ++s)
                    bf[s] = base[(s / 3 - 1) * 72 + (s % 3 - 1)];
                #pragma unroll
                for (int s = 9; s < 16; ++s) bf[s] = (_Float16)0.0f;
                c8 = wmma_f16(kstep ? a1 : a0, bf, c8);
            }
            int hy = y0 - 1 + hr;
            bool inimg = (hy >= 0 && hy < 64);
            v8h pk;                                       // 8 channels, one 16B store
            #pragma unroll
            for (int v = 0; v < 8; ++v)
                pk[v] = (_Float16)(inimg ? c8[v] : 0.0f);
            *(v8h*)&hs[hr][hc + 1][half * 8] = pk;
        }
        __syncthreads();
        #pragma unroll
        for (int k = 0; k < 4; ++k) {
            int pix = tid + 256 * k;
            int ry = pix >> 6, rc = pix & 63;
            float acc = racc[k];
            #pragma unroll 3
            for (int t = 0; t < 9; ++t) {
                v16h hv = *(const v16h*)&hs[ry + 1 + (t / 3 - 1)][rc + 1 + (t % 3 - 1)][0];
                const float* wr = &wrsT[t][chunk * 16];
                #pragma unroll
                for (int m = 0; m < 16; ++m)
                    acc += wr[m] * (float)hv[m];
            }
            racc[k] = acc;
        }
        __syncthreads();
    }
    float br0 = br[0];
    #pragma unroll
    for (int k = 0; k < 4; ++k) {
        int pix = tid + 256 * k;
        int ry = pix >> 6, rc = pix & 63;
        r_out[(size_t)b * 4096 + (y0 + ry) * 64 + rc] = racc[k] + br0;
    }
}

// ---------------------------------------------------------------------------
// Kernel 3: value iteration, one workgroup per image, fully in LDS.
// K permutation: lane-half selects r (half 0) vs v (half 1); element<9 = tap.
// ---------------------------------------------------------------------------
__global__ void vin_vi_kernel(const float* __restrict__ r_in,
                              const float* __restrict__ Wq,
                              const float* __restrict__ bq,
                              const int* __restrict__ p_in,
                              float* __restrict__ qpos_out) {
    const int b   = blockIdx.x;
    const int tid = threadIdx.x;
    const int ln  = tid & 31;
    const int wv  = __builtin_amdgcn_readfirstlane(tid >> 5);
    const int half = ln >> 4;

    __shared__ _Float16 rs[66][68];
    __shared__ _Float16 vA[66][68], vB[66][68];
    __shared__ __align__(32) _Float16 wqA[32][16];
    __shared__ float bqs[16];

    for (int idx = tid; idx < 66 * 68; idx += 256) {
        int row = idx / 68, col = idx % 68;
        int gy = row - 1, gx = col - 1;
        float v = 0.0f;
        if (gy >= 0 && gy < 64 && gx >= 0 && gx < 64)
            v = r_in[(size_t)b * 4096 + gy * 64 + gx];
        rs[row][col] = (_Float16)v;
        vA[row][col] = (_Float16)0.0f;
        vB[row][col] = (_Float16)0.0f;
    }
    for (int idx = tid; idx < 512; idx += 256) {
        int lane = idx / 16, e = idx % 16;
        int ch  = lane & 15;
        int khw = a_kidx(lane, e);
        int c   = khw >> 4;                   // 0 = r, 1 = v
        int s   = khw & 15;
        float w = 0.0f;
        if (s < 9)
            w = Wq[((ch * 2 + c) * 3 + s / 3) * 3 + s % 3];
        wqA[lane][e] = (_Float16)w;
    }
    if (tid < 16) bqs[tid] = bq[tid];
    __syncthreads();

    v16h a = *(const v16h*)&wqA[ln][0];
    v8f cinit;
    #pragma unroll
    for (int v = 0; v < 8; ++v) cinit[v] = bqs[v + half * 8];

    _Float16 (*vc)[68] = vA;
    _Float16 (*vn)[68] = vB;
    const int nlo = ln & 15;

    for (int it = 0; it < 40; ++it) {
        #pragma unroll 4
        for (int t0 = 0; t0 < 32; ++t0) {      // 256 tiles, 32 per wave
            int tl  = wv + t0 * 8;
            int pos = tl * 16 + nlo;
            int y = pos >> 6, xx = pos & 63;
            const _Float16* base = half ? &vc[y + 1][xx + 1] : &rs[y + 1][xx + 1];
            v16h bf;
            #pragma unroll
            for (int s = 0; s < 9; ++s)
                bf[s] = base[(s / 3 - 1) * 68 + (s % 3 - 1)];
            #pragma unroll
            for (int s = 9; s < 16; ++s) bf[s] = (_Float16)0.0f;
            v8f c8 = wmma_f16(a, bf, cinit);
            float m = c8[0];
            #pragma unroll
            for (int v = 1; v < 8; ++v) m = fmaxf(m, c8[v]);
            m = fmaxf(m, __shfl_xor(m, 16, 32));
            if (ln < 16) vn[y + 1][xx + 1] = (_Float16)m;
        }
        __syncthreads();
        _Float16 (*t)[68] = vc; vc = vn; vn = t;
    }

    if (tid < 16) {
        int pp = p_in[b];
        int row = pp >> 6, col = pp & 63;
        int ch = tid;
        float acc = bq[ch];
        #pragma unroll
        for (int c = 0; c < 2; ++c)
            #pragma unroll
            for (int t = 0; t < 9; ++t) {
                float src = (c == 0)
                    ? (float)rs[row + 1 + (t / 3 - 1)][col + 1 + (t % 3 - 1)]
                    : (float)vc[row + 1 + (t / 3 - 1)][col + 1 + (t % 3 - 1)];
                acc += Wq[((ch * 2 + c) * 3 + t / 3) * 3 + t % 3] * src;
            }
        qpos_out[b * 16 + ch] = acc;
    }
}

// ---------------------------------------------------------------------------
// Kernel 4: LSTM cell + FC head (tiny: VALU). One workgroup per batch row.
// out layout: [0,512) q_out ; [512,512+32768) h_new ; then c_new.
// ---------------------------------------------------------------------------
__global__ void vin_lstm_kernel(const float* __restrict__ qpos,
                                const float* __restrict__ h0,
                                const float* __restrict__ c0,
                                const float* __restrict__ W_ih,
                                const float* __restrict__ W_hh,
                                const float* __restrict__ b_ih,
                                const float* __restrict__ b_hh,
                                const float* __restrict__ W_fc,
                                float* __restrict__ out) {
    const int b = blockIdx.x, tid = threadIdx.x;
    __shared__ float h0s[512], qs[16], zs[2048], rh[512], qa[8];
    for (int k = tid; k < 512; k += 256) h0s[k] = h0[b * 512 + k];
    if (tid < 16) qs[tid] = qpos[b * 16 + tid];
    if (tid < 8) qa[tid] = 0.0f;
    __syncthreads();
    for (int j = tid; j < 2048; j += 256) {
        float acc = b_ih[j] + b_hh[j];
        const float* wi = W_ih + j * 16;
        #pragma unroll
        for (int k = 0; k < 16; ++k) acc += qs[k] * wi[k];
        const float* wh = W_hh + (size_t)j * 512;
        for (int k = 0; k < 512; ++k) acc += h0s[k] * wh[k];
        zs[j] = acc;
    }
    __syncthreads();
    for (int k = tid; k < 512; k += 256) {
        float i = sigf(zs[k]);
        float f = sigf(zs[512 + k]);
        float g = tanhf(zs[1024 + k]);
        float o = sigf(zs[1536 + k]);
        float cn = f * c0[b * 512 + k] + i * g;
        float hn = o * tanhf(cn);
        out[512 + (size_t)b * 512 + k] = hn;
        out[512 + 32768 + (size_t)b * 512 + k] = cn;
        rh[k] = fmaxf(hn, 0.0f);
    }
    __syncthreads();
    if (tid < 64) {
        int a = tid >> 3, part = tid & 7;
        float acc = 0.0f;
        const float* wf = W_fc + a * 512 + part * 64;
        const float* rr = rh + part * 64;
        #pragma unroll
        for (int k = 0; k < 64; ++k) acc += rr[k] * wf[k];
        atomicAdd(&qa[a], acc);
    }
    __syncthreads();
    if (tid < 8) out[(size_t)b * 8 + tid] = qa[tid];
}

// ---------------------------------------------------------------------------
extern "C" void kernel_launch(void* const* d_in, const int* in_sizes, int n_in,
                              void* d_out, int out_size, void* d_ws, size_t ws_size,
                              hipStream_t stream) {
    const float* x   = (const float*)d_in[0];
    const float* h0  = (const float*)d_in[1];
    const float* c0  = (const float*)d_in[2];
    const float* Wh  = (const float*)d_in[3];
    const float* bh  = (const float*)d_in[4];
    const float* Wr  = (const float*)d_in[5];
    const float* br  = (const float*)d_in[6];
    const float* Wq  = (const float*)d_in[7];
    const float* bq  = (const float*)d_in[8];
    const float* Wih = (const float*)d_in[9];
    const float* Whh = (const float*)d_in[10];
    const float* bih = (const float*)d_in[11];
    const float* bhh = (const float*)d_in[12];
    const float* Wfc = (const float*)d_in[13];
    float* out = (float*)d_out;

    char* ws = (char*)d_ws;
    int*   p_ws = (int*)ws;
    float* r_ws = (float*)(ws + 1024);
    float* q_ws = (float*)(ws + 1024 + (size_t)64 * 4096 * 4);

    vin_pos_kernel <<<64,          256, 0, stream>>>(x, p_ws);
    vin_hr_kernel  <<<dim3(4, 64), 256, 0, stream>>>(x, Wh, bh, Wr, br, r_ws);
    vin_vi_kernel  <<<64,          256, 0, stream>>>(r_ws, Wq, bq, p_ws, q_ws);
    vin_lstm_kernel<<<64,          256, 0, stream>>>(q_ws, h0, c0, Wih, Whh, bih, bhh, Wfc, out);
}
